// PINN_Ncl_50766513438737
// MI455X (gfx1250) — matrix operand order repaired and checked by reading, not compiled, not run on hardware
//
#include <hip/hip_runtime.h>
#include <math.h>

typedef __attribute__((ext_vector_type(16))) _Float16 v16h;
typedef __attribute__((ext_vector_type(8)))  _Float16 v8h;
typedef __attribute__((ext_vector_type(8)))  float    v8f;

#define WAVES_PB 8
// LDS byte offsets
#define OFF_DW1   0        // f16 [64][64] div W1 row-major [out][in]
#define OFF_DW2   8192     // f16 [64][64]
#define OFF_DW1T  16384    // f16 [64][64] transposed [in][out]
#define OFF_DW2T  24576
#define OFF_MW1   32768
#define OFF_MW2   40960
#define OFF_DW0   49152    // f16 [64][4] (padded fan_in 3->4)
#define OFF_MW0   49664
#define OFF_MW3   50176    // f16 [16][64] (rows 3..15 zero)
#define OFF_DW3   52224    // f32 [64]
#define OFF_DB    52480    // f32 [3][64]
#define OFF_MB    53248    // f32 [3][64]
#define OFF_MB3   54016    // f32 [4]
#define OFF_WAVE  54144    // per-wave scratch: 8192 B each (act 2KB + 3x sig 2KB)
#define SMEM_BYTES (OFF_WAVE + WAVES_PB * 8192)

#define LOG2E_F 1.4426950408889634f
#define LN2_F   0.6931471805599453f

__device__ __forceinline__ v8f wmma_f16(v16h a, v16h b, v8f c) {
  return __builtin_amdgcn_wmma_f32_16x16x32_f16(false, a, false, b, (short)0, c,
                                                false, false);
}

__device__ __forceinline__ float fast_rcp(float x) {
  return __builtin_amdgcn_rcpf(x);    // v_rcp_f32
}
__device__ __forceinline__ float fast_exp2(float x) {
  return __builtin_amdgcn_exp2f(x);   // v_exp_f32 (2^x), no range guards
}
__device__ __forceinline__ float fast_log2(float x) {
  return __builtin_amdgcn_logf(x);    // v_log_f32 (log2 x), no range guards
}

// A fragment (16x32 f16): lanes 0-15 = M rows; v[0..3]=K(kbase + laneHi*8 ..+7),
// v[4..7]=K(kbase+16+laneHi*8 ..). act is [16 pts][64 feats] f16 row-major.
__device__ __forceinline__ v16h load_a_frag(const _Float16* act, int laneLo,
                                            int laneHi, int kbase) {
  const _Float16* p = act + laneLo * 64 + kbase + laneHi * 8;
  v8h lo = *(const v8h*)(p);
  v8h hi = *(const v8h*)(p + 16);
  v16h a;
#pragma unroll
  for (int i = 0; i < 8; ++i) { a[i] = lo[i]; a[i + 8] = hi[i]; }
  return a;
}

// B fragment (32x16 f16): lane column N = ntile*16+laneLo; v[0..3]=K(kbase + laneHi*8 ..+7),
// v[4..7]=K(kbase+16+laneHi*8 ..) (same order as A fragment).
// W is row-major [64 rows][64] f16, row index = output column of the GEMM.
__device__ __forceinline__ v16h load_b_frag(const _Float16* W, int ntile,
                                            int laneLo, int laneHi, int kbase) {
  const _Float16* p = W + (ntile * 16 + laneLo) * 64 + kbase + laneHi * 8;
  v8h lo = *(const v8h*)(p);
  v8h hi = *(const v8h*)(p + 16);
  v16h b;
#pragma unroll
  for (int i = 0; i < 8; ++i) { b[i] = lo[i]; b[i + 8] = hi[i]; }
  return b;
}

// softplus20(z) = log1p(exp(20z))/20 (stable form), s = sigmoid(20z).
// Uses raw v_exp/v_log/v_rcp: log arg is 1+e in [1,2] (never denormal),
// exp2 underflow -> 0 is the desired tail behavior.
__device__ __forceinline__ void softplus20(float z, float& h, float& s) {
  float t = 20.0f * z;
  float e = fast_exp2(-20.0f * LOG2E_F * fabsf(z));   // exp(-|t|)
  float ope = 1.0f + e;
  float l2 = fast_log2(ope);                           // log2(1+e) in [0,1]
  h = fmaxf(t, 0.0f) * 0.05f + l2 * (LN2_F * 0.05f);
  float r = fast_rcp(ope);
  s = (t >= 0.0f) ? r : (1.0f - r);
}

// tanh(z) = copysign((1-e)/(1+e), z), e = exp(-2|z|) via exp2 + rcp.
__device__ __forceinline__ float fast_tanh(float z) {
  float e = fast_exp2(-2.0f * LOG2E_F * fabsf(z));
  float r = (1.0f - e) * fast_rcp(1.0f + e);
  return copysignf(r, z);
}

// One hidden layer: act[16x64] <- activation(act @ W^T + b). SP=softplus20
// (also stores sigmoid(20z) to sigL in C-register order), else tanh.
template <bool SP>
__device__ __forceinline__ void hidden_layer(const _Float16* W, const float* bias,
                                             _Float16* act, _Float16* sigL,
                                             int laneLo, int laneHi, int lane) {
  v16h A0 = load_a_frag(act, laneLo, laneHi, 0);
  v16h A1 = load_a_frag(act, laneLo, laneHi, 32);
#pragma unroll
  for (int t = 0; t < 4; ++t) {
    v16h B0 = load_b_frag(W, t, laneLo, laneHi, 0);
    v16h B1 = load_b_frag(W, t, laneLo, laneHi, 32);
    v8f c = {};
    c = wmma_f16(A0, B0, c);
    c = wmma_f16(A1, B1, c);
    float bn = bias[t * 16 + laneLo];
    v8h sv;
#pragma unroll
    for (int r = 0; r < 8; ++r) {
      float z = c[r] + bn;
      float h;
      if (SP) { float s; softplus20(z, h, s); sv[r] = (_Float16)s; }
      else    { h = fast_tanh(z); }
      act[(r + laneHi * 8) * 64 + t * 16 + laneLo] = (_Float16)h;
    }
    if (SP) *(v8h*)(sigL + t * 256 + lane * 8) = sv;
  }
}

// Input layer: K=3 padded into a single K=32 step. W0 is f16 [64][4].
template <bool SP>
__device__ __forceinline__ void input_layer(const float* tx, int pbase,
                                            const _Float16* W0, const float* bias,
                                            _Float16* act, _Float16* sigL,
                                            int laneLo, int laneHi, int lane) {
  v16h A = {};
  if (laneHi == 0) {  // lanes 0-15 hold K=0..7; K>=3 stays zero
    const float* xp = tx + (size_t)(pbase + laneLo) * 3;
    A[0] = (_Float16)xp[0];
    A[1] = (_Float16)xp[1];
    A[2] = (_Float16)xp[2];
  }
#pragma unroll
  for (int t = 0; t < 4; ++t) {
    v16h B = {};
    if (laneHi == 0) {
      const _Float16* w = W0 + (t * 16 + laneLo) * 4;
      B[0] = w[0]; B[1] = w[1]; B[2] = w[2];
    }
    v8f c = {};
    c = wmma_f16(A, B, c);
    float bn = bias[t * 16 + laneLo];
    v8h sv;
#pragma unroll
    for (int r = 0; r < 8; ++r) {
      float z = c[r] + bn;
      float h;
      if (SP) { float s; softplus20(z, h, s); sv[r] = (_Float16)s; }
      else    { h = fast_tanh(z); }
      act[(r + laneHi * 8) * 64 + t * 16 + laneLo] = (_Float16)h;
    }
    if (SP) *(v8h*)(sigL + t * 256 + lane * 8) = sv;
  }
}

__global__ void __launch_bounds__(256)
pinn_wmma_kernel(const float* __restrict__ tx,
                 const float* __restrict__ g_dW0, const float* __restrict__ g_db0,
                 const float* __restrict__ g_dW1, const float* __restrict__ g_db1,
                 const float* __restrict__ g_dW2, const float* __restrict__ g_db2,
                 const float* __restrict__ g_dW3,
                 const float* __restrict__ g_mW0, const float* __restrict__ g_mb0,
                 const float* __restrict__ g_mW1, const float* __restrict__ g_mb1,
                 const float* __restrict__ g_mW2, const float* __restrict__ g_mb2,
                 const float* __restrict__ g_mW3, const float* __restrict__ g_mb3,
                 float* __restrict__ out) {
  extern __shared__ __align__(16) char smem[];
  _Float16* dW1  = (_Float16*)(smem + OFF_DW1);
  _Float16* dW2  = (_Float16*)(smem + OFF_DW2);
  _Float16* dW1T = (_Float16*)(smem + OFF_DW1T);
  _Float16* dW2T = (_Float16*)(smem + OFF_DW2T);
  _Float16* mW1  = (_Float16*)(smem + OFF_MW1);
  _Float16* mW2  = (_Float16*)(smem + OFF_MW2);
  _Float16* dW0  = (_Float16*)(smem + OFF_DW0);
  _Float16* mW0  = (_Float16*)(smem + OFF_MW0);
  _Float16* mW3  = (_Float16*)(smem + OFF_MW3);
  float*    dW3f = (float*)(smem + OFF_DW3);
  float*    dB   = (float*)(smem + OFF_DB);
  float*    mB   = (float*)(smem + OFF_MB);
  float*    mb3  = (float*)(smem + OFF_MB3);

  const int tid = threadIdx.x;
  // ---- cooperative weight conversion fp32 -> f16 LDS ----
  for (int i = tid; i < 4096; i += 256) {
    float w1 = g_dW1[i], w2 = g_dW2[i];
    dW1[i] = (_Float16)w1;
    dW2[i] = (_Float16)w2;
    int r = i >> 6, c = i & 63;
    dW1T[c * 64 + r] = (_Float16)w1;
    dW2T[c * 64 + r] = (_Float16)w2;
    mW1[i] = (_Float16)g_mW1[i];
    mW2[i] = (_Float16)g_mW2[i];
  }
  for (int i = tid; i < 1024; i += 256) {
    int r = i >> 6, c = i & 63;
    mW3[i] = (r < 3) ? (_Float16)g_mW3[r * 64 + c] : (_Float16)0.0f;
  }
  if (tid < 64) {
    dW0[tid * 4 + 0] = (_Float16)g_dW0[tid * 3 + 0];
    dW0[tid * 4 + 1] = (_Float16)g_dW0[tid * 3 + 1];
    dW0[tid * 4 + 2] = (_Float16)g_dW0[tid * 3 + 2];
    dW0[tid * 4 + 3] = (_Float16)0.0f;
    mW0[tid * 4 + 0] = (_Float16)g_mW0[tid * 3 + 0];
    mW0[tid * 4 + 1] = (_Float16)g_mW0[tid * 3 + 1];
    mW0[tid * 4 + 2] = (_Float16)g_mW0[tid * 3 + 2];
    mW0[tid * 4 + 3] = (_Float16)0.0f;
    dW3f[tid] = g_dW3[tid];
    dB[tid] = g_db0[tid]; dB[64 + tid] = g_db1[tid]; dB[128 + tid] = g_db2[tid];
    mB[tid] = g_mb0[tid]; mB[64 + tid] = g_mb1[tid]; mB[128 + tid] = g_mb2[tid];
  }
  if (tid < 4) mb3[tid] = (tid < 3) ? g_mb3[tid] : 0.0f;
  __syncthreads();

  const int wave = tid >> 5, lane = tid & 31;
  const int laneLo = lane & 15, laneHi = lane >> 4;
  const int pbase = (blockIdx.x * WAVES_PB + wave) * 16;

  _Float16* wb  = (_Float16*)(smem + OFF_WAVE) + wave * 4096;
  _Float16* act = wb;          // 16x64 f16
  _Float16* sg0 = wb + 1024;   // sigmoid(20 z_l), C-register order, per layer
  _Float16* sg1 = wb + 2048;
  _Float16* sg2 = wb + 3072;

  // ================= div net forward (softplus20, cache sigmoids) =========
  input_layer<true>(tx, pbase, dW0, dB, act, sg0, laneLo, laneHi, lane);
  hidden_layer<true>(dW1, dB + 64,  act, sg1, laneLo, laneHi, lane);
  hidden_layer<true>(dW2, dB + 128, act, sg2, laneLo, laneHi, lane);

  // ================= div net backward ======================================
  // dz2[p][i] = W3[i] * sig2[p][i]  (ds/dh3 is just the W3 row)
#pragma unroll
  for (int t = 0; t < 4; ++t) {
    v8h sv = *(const v8h*)(sg2 + t * 256 + lane * 8);
    float w3 = dW3f[t * 16 + laneLo];
#pragma unroll
    for (int r = 0; r < 8; ++r)
      act[(r + laneHi * 8) * 64 + t * 16 + laneLo] = (_Float16)((float)sv[r] * w3);
  }
  // dz1 = (dz2 @ W2) * sig1   (B from W2^T -> contiguous fragments)
  {
    v16h A0 = load_a_frag(act, laneLo, laneHi, 0);
    v16h A1 = load_a_frag(act, laneLo, laneHi, 32);
#pragma unroll
    for (int t = 0; t < 4; ++t) {
      v16h B0 = load_b_frag(dW2T, t, laneLo, laneHi, 0);
      v16h B1 = load_b_frag(dW2T, t, laneLo, laneHi, 32);
      v8f c = {};
      c = wmma_f16(A0, B0, c);
      c = wmma_f16(A1, B1, c);
      v8h sv = *(const v8h*)(sg1 + t * 256 + lane * 8);
#pragma unroll
      for (int r = 0; r < 8; ++r)
        act[(r + laneHi * 8) * 64 + t * 16 + laneLo] =
            (_Float16)(c[r] * (float)sv[r]);
    }
  }
  // dz0 = (dz1 @ W1) * sig0 ; g[p][1:3] = dz0 @ W0[:,1:3] via lane reduction
  {
    v16h A0 = load_a_frag(act, laneLo, laneHi, 0);
    v16h A1 = load_a_frag(act, laneLo, laneHi, 32);
    float ax[8], ay[8];
#pragma unroll
    for (int r = 0; r < 8; ++r) { ax[r] = 0.0f; ay[r] = 0.0f; }
#pragma unroll
    for (int t = 0; t < 4; ++t) {
      v16h B0 = load_b_frag(dW1T, t, laneLo, laneHi, 0);
      v16h B1 = load_b_frag(dW1T, t, laneLo, laneHi, 32);
      v8f c = {};
      c = wmma_f16(A0, B0, c);
      c = wmma_f16(A1, B1, c);
      v8h sv = *(const v8h*)(sg0 + t * 256 + lane * 8);
      float wx = (float)dW0[(t * 16 + laneLo) * 4 + 1];
      float wy = (float)dW0[(t * 16 + laneLo) * 4 + 2];
#pragma unroll
      for (int r = 0; r < 8; ++r) {
        float dz = c[r] * (float)sv[r];
        ax[r] += dz * wx;
        ay[r] += dz * wy;
      }
    }
#pragma unroll
    for (int m = 8; m >= 1; m >>= 1) {
#pragma unroll
      for (int r = 0; r < 8; ++r) {
        ax[r] += __shfl_xor(ax[r], m, 32);
        ay[r] += __shfl_xor(ay[r], m, 32);
      }
    }
    if (laneLo == 0) {
#pragma unroll
      for (int r = 0; r < 8; ++r) {
        size_t p = (size_t)(pbase + r + laneHi * 8);
        out[p * 5 + 3] =  ay[r];   // ds/dy
        out[p * 5 + 4] = -ax[r];   // -ds/dx
      }
    }
  }

  // ================= mom net forward (tanh) ================================
  input_layer<false>(tx, pbase, mW0, mB, act, nullptr, laneLo, laneHi, lane);
  hidden_layer<false>(mW1, mB + 64,  act, nullptr, laneLo, laneHi, lane);
  hidden_layer<false>(mW2, mB + 128, act, nullptr, laneLo, laneHi, lane);
  {
    v16h A0 = load_a_frag(act, laneLo, laneHi, 0);
    v16h A1 = load_a_frag(act, laneLo, laneHi, 32);
    v16h B0 = load_b_frag(mW3, 0, laneLo, laneHi, 0);
    v16h B1 = load_b_frag(mW3, 0, laneLo, laneHi, 32);
    v8f c = {};
    c = wmma_f16(A0, B0, c);
    c = wmma_f16(A1, B1, c);
    if (laneLo < 3) {
      float b3 = mb3[laneLo];
#pragma unroll
      for (int r = 0; r < 8; ++r) {
        size_t p = (size_t)(pbase + r + laneHi * 8);
        out[p * 5 + laneLo] = c[r] + b3;
      }
    }
  }
}

extern "C" void kernel_launch(void* const* d_in, const int* in_sizes, int n_in,
                              void* d_out, int out_size, void* d_ws, size_t ws_size,
                              hipStream_t stream) {
  (void)n_in; (void)out_size; (void)d_ws; (void)ws_size;
  const float* tx  = (const float*)d_in[0];
  const float* dW0 = (const float*)d_in[1];
  const float* db0 = (const float*)d_in[2];
  const float* dW1 = (const float*)d_in[3];
  const float* db1 = (const float*)d_in[4];
  const float* dW2 = (const float*)d_in[5];
  const float* db2 = (const float*)d_in[6];
  const float* dW3 = (const float*)d_in[7];
  // d_in[8] = div_b3: unused (output depends only on gradient of s)
  const float* mW0 = (const float*)d_in[9];
  const float* mb0 = (const float*)d_in[10];
  const float* mW1 = (const float*)d_in[11];
  const float* mb1 = (const float*)d_in[12];
  const float* mW2 = (const float*)d_in[13];
  const float* mb2 = (const float*)d_in[14];
  const float* mW3 = (const float*)d_in[15];
  const float* mb3 = (const float*)d_in[16];
  float* out = (float*)d_out;

  int npts = in_sizes[0] / 3;            // 524288
  int blocks = npts / (WAVES_PB * 16);   // 128 points per block
  pinn_wmma_kernel<<<blocks, 256, SMEM_BYTES, stream>>>(
      tx, dW0, db0, dW1, db1, dW2, db2, dW3,
      mW0, mb0, mW1, mb1, mW2, mb2, mW3, mb3, out);
}
